// GLMBlockOpt_25031069401203
// MI455X (gfx1250) — matrix operand order repaired and checked by reading, not compiled, not run on hardware
//
#include <hip/hip_runtime.h>
#include <math.h>

// ---------------------------------------------------------------------------
// GLM block on MI455X (gfx1250): bf16 WMMA GEMMs + double-buffered
// async-to-LDS staging, 2 tiles in flight, branch-free steady-state loop.
// ---------------------------------------------------------------------------

typedef __bf16 bf16;
typedef __attribute__((ext_vector_type(16))) __bf16 v16bf;
typedef __attribute__((ext_vector_type(8)))  __bf16 v8bf;
typedef __attribute__((ext_vector_type(8)))  float  v8f;
typedef __attribute__((ext_vector_type(4)))  int    v4i;

#if defined(__AMDGCN__) && __has_builtin(__builtin_amdgcn_global_load_async_to_lds_b128) && __has_builtin(__builtin_amdgcn_s_wait_asynccnt)
#define HAS_ASYNC_LDS 1
#else
#define HAS_ASYNC_LDS 0
#endif

__device__ __forceinline__ void async_copy_b128(const void* g, void* l) {
#if HAS_ASYNC_LDS
  __builtin_amdgcn_global_load_async_to_lds_b128(
      (__attribute__((address_space(1))) v4i*)g,
      (__attribute__((address_space(3))) v4i*)l, 0, 0);
#else
  *(v4i*)l = *(const v4i*)g;
#endif
}

__device__ __forceinline__ void wait_async_le4() {
#if HAS_ASYNC_LDS
  __builtin_amdgcn_s_wait_asynccnt(4);
#endif
}
__device__ __forceinline__ void wait_async_le0() {
#if HAS_ASYNC_LDS
  __builtin_amdgcn_s_wait_asynccnt(0);
#endif
}

// ---------------------------------------------------------------------------
// Tiled WMMA GEMM: C[M,N] = alpha * A[M,K] (row-major) x B[N,K]^T (row-major)
// M = gridDim.y*128, N = gridDim.x*128; all dims multiples of tile sizes;
// K/32 must be >= 2 (true for every GEMM in this block).
// EPI: 0 = f32 store, 1 = bf16 store, 2 = f32 residual+acc, 3 = f32 residual+bias+acc
// ---------------------------------------------------------------------------
#define BM 128
#define BN 128
#define BK 32

template<int EPI>
__global__ __launch_bounds__(256)
void gemm_bf16_wmma_kernel(const bf16* __restrict__ A, int lda,
                           const bf16* __restrict__ B, int ldb,
                           float* __restrict__ Cf, bf16* __restrict__ Cb, int ldc,
                           const float* __restrict__ residual,
                           const float* __restrict__ bias,
                           int K, float alpha)
{
  __shared__ alignas(64) bf16 sA[2][BM * BK];
  __shared__ alignas(64) bf16 sB[2][BN * BK];

  const int m0   = blockIdx.y * BM;
  const int n0   = blockIdx.x * BN;
  const int tid  = threadIdx.x;
  const int wid  = tid >> 5;         // 8 waves (wave32)
  const int lane = tid & 31;

  const int wm = (wid >> 2) * 64;    // wave M offset: 0 / 64
  const int wn = (wid & 3) * 32;     // wave N offset: 0..96

  const v8f vzero = {0.f, 0.f, 0.f, 0.f, 0.f, 0.f, 0.f, 0.f};
  v8f acc[4][2];
#pragma unroll
  for (int i = 0; i < 4; ++i)
#pragma unroll
    for (int j = 0; j < 2; ++j) acc[i][j] = vzero;

  // per-thread global->LDS copy: 256 thr * 2 iters * 8 bf16 = 128x32 tile
  const int crow = tid >> 2;          // 0..63
  const int ccol = (tid & 3) * 8;     // 0,8,16,24

  // Issue the 4 async b128 copies of one K-tile into LDS buffer b.
  auto issue_tile = [&](int kt, int b) {
    const int k0 = kt * BK;
#pragma unroll
    for (int it = 0; it < 2; ++it) {
      const int r = crow + it * 64;
      async_copy_b128(&A[(size_t)(m0 + r) * lda + k0 + ccol], &sA[b][r * BK + ccol]);
      async_copy_b128(&B[(size_t)(n0 + r) * ldb + k0 + ccol], &sB[b][r * BK + ccol]);
    }
  };

  // 8x v_wmma_f32_16x16x32_bf16 on one LDS buffer.
  auto compute_tile = [&](int b) {
    // B fragments: lane holds 16 consecutive K of column n (contiguous in LDS)
    v16bf bfrag[2];
#pragma unroll
    for (int j = 0; j < 2; ++j) {
      const int n  = wn + j * 16 + (lane & 15);
      const int kb = (lane >> 4) * 16;
      bfrag[j] = *(const v16bf*)&sB[b][n * BK + kb];
    }
#pragma unroll
    for (int i = 0; i < 4; ++i) {
      const int m  = wm + i * 16 + (lane & 15);
      const int kb = (lane >> 4) * 16;
      const v8bf lo = *(const v8bf*)&sA[b][m * BK + kb];        // K = kb..kb+7
      const v8bf hi = *(const v8bf*)&sA[b][m * BK + kb + 8];    // K = kb+8..kb+15
      const v16bf afrag = __builtin_shufflevector(lo, hi,
          0, 1, 2, 3, 4, 5, 6, 7, 8, 9, 10, 11, 12, 13, 14, 15);
#pragma unroll
      for (int j = 0; j < 2; ++j) {
        acc[i][j] = __builtin_amdgcn_wmma_f32_16x16x32_bf16(
            false, afrag, false, bfrag[j], (short)0, acc[i][j], false, false);
      }
    }
  };

  const int nk = K / BK;

  // Prologue: two tiles in flight (8 outstanding async copies per thread).
  issue_tile(0, 0);
  if (nk > 1) issue_tile(1, 1);

  // Steady state: unconditional s_wait_asynccnt 4 — in-order async retirement
  // guarantees the oldest 4 copies (current tile) are complete while the next
  // tile's 4 remain in flight.  Tile kt+2 is issued only after the trailing
  // barrier proves all waves finished reading this buffer.
#pragma unroll 2
  for (int kt = 0; kt < nk - 1; ++kt) {
    const int buf = kt & 1;
    wait_async_le4();
    __syncthreads();
    compute_tile(buf);
    __syncthreads();
    if (kt + 2 < nk) issue_tile(kt + 2, buf);
  }

  // Peeled final K-step: drain everything.
  wait_async_le0();
  __syncthreads();
  compute_tile((nk - 1) & 1);

  // Epilogue: C 16x16 frag layout -> row = base + vgpr + (lane>=16)*8, col = lane&15
#pragma unroll
  for (int i = 0; i < 4; ++i) {
#pragma unroll
    for (int j = 0; j < 2; ++j) {
      const int col   = n0 + wn + j * 16 + (lane & 15);
      const int rbase = m0 + wm + i * 16 + ((lane >> 4) * 8);
#pragma unroll
      for (int r = 0; r < 8; ++r) {
        const int row = rbase + r;
        const float v = acc[i][j][r] * alpha;
        const size_t idx = (size_t)row * ldc + col;
        if (EPI == 0)      Cf[idx] = v;
        else if (EPI == 1) Cb[idx] = (bf16)v;
        else if (EPI == 2) Cf[idx] = residual[idx] + v;
        else               Cf[idx] = residual[idx] + bias[col] + v;
      }
    }
  }
}

// ---------------------------------------------------------------------------
// RMSNorm (f32 in, bf16 out), one block per row.
// ---------------------------------------------------------------------------
__global__ __launch_bounds__(256)
void rmsnorm_bf16_kernel(const float* __restrict__ x, const float* __restrict__ w,
                         bf16* __restrict__ out, int H)
{
  const int s = blockIdx.x;
  const float* row = x + (size_t)s * H;
  bf16* orow = out + (size_t)s * H;
  __shared__ float red[256];
  const int tid = threadIdx.x;
  float ss = 0.f;
  for (int i = tid; i < H; i += 256) { const float v = row[i]; ss += v * v; }
  red[tid] = ss; __syncthreads();
  for (int s2 = 128; s2 > 0; s2 >>= 1) {
    if (tid < s2) red[tid] += red[tid + s2];
    __syncthreads();
  }
  const float inv = rsqrtf(red[0] / (float)H + 1e-6f);
  for (int i = tid; i < H; i += 256) orow[i] = (bf16)(row[i] * inv * w[i]);
}

// ---------------------------------------------------------------------------
// Causal softmax over one score row (f32 in, bf16 out).
// ---------------------------------------------------------------------------
__global__ __launch_bounds__(256)
void softmax_causal_kernel(const float* __restrict__ scores, bf16* __restrict__ probs, int S)
{
  const int q = blockIdx.x;
  const float* row = scores + (size_t)q * S;
  bf16* prow = probs + (size_t)q * S;
  const int tid = threadIdx.x;
  __shared__ float red[256];

  float mx = -1e30f;
  for (int k = tid; k <= q; k += 256) mx = fmaxf(mx, row[k]);
  red[tid] = mx; __syncthreads();
  for (int s2 = 128; s2 > 0; s2 >>= 1) {
    if (tid < s2) red[tid] = fmaxf(red[tid], red[tid + s2]);
    __syncthreads();
  }
  mx = red[0]; __syncthreads();

  float sum = 0.f;
  for (int k = tid; k <= q; k += 256) sum += __expf(row[k] - mx);
  red[tid] = sum; __syncthreads();
  for (int s2 = 128; s2 > 0; s2 >>= 1) {
    if (tid < s2) red[tid] += red[tid + s2];
    __syncthreads();
  }
  const float inv = 1.0f / red[0];
  for (int k = tid; k < S; k += 256)
    prow[k] = (k <= q) ? (bf16)(__expf(row[k] - mx) * inv) : (bf16)0.f;
}

// ---------------------------------------------------------------------------
// f32 -> bf16 conversion (grid-stride).
// ---------------------------------------------------------------------------
__global__ void f32_to_bf16_kernel(const float* __restrict__ in, bf16* __restrict__ out, size_t n)
{
  size_t i = (size_t)blockIdx.x * blockDim.x + threadIdx.x;
  const size_t stride = (size_t)gridDim.x * blockDim.x;
  for (; i < n; i += stride) out[i] = (bf16)in[i];
}

// ---------------------------------------------------------------------------
// Split qkv [S,3H] f32 into bf16 q[NH,S,HD], k[NH,S,HD], vT[NH,HD,S].
// ---------------------------------------------------------------------------
__global__ void qkv_split_kernel(const float* __restrict__ qkv,
                                 bf16* __restrict__ qb, bf16* __restrict__ kb,
                                 bf16* __restrict__ vT,
                                 int S, int H, int HD)
{
  const int idx = blockIdx.x * blockDim.x + threadIdx.x;   // S*H
  if (idx >= S * H) return;
  const int s = idx / H, c = idx % H;
  const int h = c / HD, d = c % HD;
  const float* base = qkv + (size_t)s * 3 * H;
  qb[(size_t)h * S * HD + (size_t)s * HD + d] = (bf16)base[c];
  kb[(size_t)h * S * HD + (size_t)s * HD + d] = (bf16)base[H + c];
  vT[(size_t)h * HD * S + (size_t)d * S + s] = (bf16)base[2 * H + c];
}

// ---------------------------------------------------------------------------
// int4 dequant: qweight[H, FF/2] byte-valued ints -> bf16 w_deq[H, FF].
// ---------------------------------------------------------------------------
__global__ void dequant_int4_kernel(const int* __restrict__ qweight,
                                    const int* __restrict__ qzeros,
                                    const float* __restrict__ scales,
                                    bf16* __restrict__ wdeq,
                                    int H, int FF, int GS)
{
  const int idx = blockIdx.x * blockDim.x + threadIdx.x;   // H * FF/2
  const int half = FF / 2;
  if (idx >= H * half) return;
  const int row = idx / half, c2 = idx % half;
  const int q = qweight[idx];
  const int col = c2 * 2;
  const int g = col / GS;
  const int ng = FF / GS;
  const float z = (float)qzeros[row * ng + g];
  const float s = scales[row * ng + g];
  wdeq[(size_t)row * FF + col]     = (bf16)(((float)((q >> 4) & 15) - z) * s);
  wdeq[(size_t)row * FF + col + 1] = (bf16)(((float)(q & 15) - z) * s);
}

// ---------------------------------------------------------------------------
// SwiGLU: act[s,f] = silu(inter[s,f]) * inter[s,FF+f]  (f32 in, bf16 out).
// ---------------------------------------------------------------------------
__global__ void swiglu_kernel(const float* __restrict__ inter, bf16* __restrict__ act,
                              int S, int FF)
{
  size_t i = (size_t)blockIdx.x * blockDim.x + threadIdx.x;
  const size_t total = (size_t)S * FF;
  const size_t stride = (size_t)gridDim.x * blockDim.x;
  for (; i < total; i += stride) {
    const size_t s = i / FF, f = i % FF;
    const float g = inter[s * (size_t)(2 * FF) + f];
    const float u = inter[s * (size_t)(2 * FF) + FF + f];
    act[i] = (bf16)((g / (1.f + __expf(-g))) * u);
  }
}

// ---------------------------------------------------------------------------
// Launch sequence.
// ---------------------------------------------------------------------------
extern "C" void kernel_launch(void* const* d_in, const int* in_sizes, int n_in,
                              void* d_out, int out_size, void* d_ws, size_t ws_size,
                              hipStream_t stream)
{
  (void)in_sizes; (void)n_in; (void)out_size; (void)ws_size;
  constexpr int S = 2048, H = 4096, NH = 32, HD = 128, FF = 13696, GS = 128;
  constexpr int H3 = 3 * H, FF2 = 2 * FF;
  const float SCALE = 0.08838834764831845f;  // 1/sqrt(128)

  const float* hidden     = (const float*)d_in[0];
  const float* w_norm1    = (const float*)d_in[1];
  const float* w_norm2    = (const float*)d_in[2];
  const float* w_qkv      = (const float*)d_in[3];
  const float* w_attn_out = (const float*)d_in[4];
  const float* w_h4h      = (const float*)d_in[5];
  const float* scales     = (const float*)d_in[6];
  const float* bias_4h    = (const float*)d_in[7];
  const int*   qweight    = (const int*)d_in[8];
  const int*   qzeros     = (const int*)d_in[9];
  float* out = (float*)d_out;

  char* ws = (char*)d_ws;
  size_t off = 0;
  auto alloc = [&](size_t bytes) -> void* {
    void* p = ws + off;
    off += (bytes + 255) & ~(size_t)255;
    return p;
  };

  // Persistent region.
  bf16*  w_h4h_bf = (bf16*)alloc((size_t)FF2 * H * 2);
  bf16*  wdeq_bf  = (bf16*)alloc((size_t)H * FF * 2);
  bf16*  ln_bf    = (bf16*)alloc((size_t)S * H * 2);      // ln1, then ln2
  float* x1       = (float*)alloc((size_t)S * H * 4);
  bf16*  act_bf   = (bf16*)alloc((size_t)S * FF * 2);
  bf16*  ctx_bf   = (bf16*)alloc((size_t)S * H * 2);

  // Attention scratch zone, aliased later by MLP `inter` (all dead by then).
  const size_t zbase = off;
  bf16*  wqkv_bf  = (bf16*)alloc((size_t)H3 * H * 2);
  bf16*  wattn_bf = (bf16*)alloc((size_t)H * H * 2);
  float* qkvf     = (float*)alloc((size_t)S * H3 * 4);
  bf16*  qb       = (bf16*)alloc((size_t)NH * S * HD * 2);
  bf16*  kb       = (bf16*)alloc((size_t)NH * S * HD * 2);
  bf16*  vT       = (bf16*)alloc((size_t)NH * HD * S * 2);
  float* scoresb  = (float*)alloc((size_t)S * S * 4);
  bf16*  probsb   = (bf16*)alloc((size_t)S * S * 2);
  float* inter    = (float*)(ws + zbase);                  // [S, 2*FF] alias

  const dim3 blk(256);

  // Weight conversions + dequant.
  f32_to_bf16_kernel<<<4096, blk, 0, stream>>>(w_qkv, wqkv_bf, (size_t)H3 * H);
  f32_to_bf16_kernel<<<4096, blk, 0, stream>>>(w_attn_out, wattn_bf, (size_t)H * H);
  f32_to_bf16_kernel<<<8192, blk, 0, stream>>>(w_h4h, w_h4h_bf, (size_t)FF2 * H);
  {
    const int total = H * (FF / 2);
    dequant_int4_kernel<<<(total + 255) / 256, blk, 0, stream>>>(
        qweight, qzeros, scales, wdeq_bf, H, FF, GS);
  }

  // ln1 = rmsnorm(x) ; qkv = ln1 @ w_qkv^T
  rmsnorm_bf16_kernel<<<S, blk, 0, stream>>>(hidden, w_norm1, ln_bf, H);
  gemm_bf16_wmma_kernel<0><<<dim3(H3 / BN, S / BM), blk, 0, stream>>>(
      ln_bf, H, wqkv_bf, H, qkvf, nullptr, H3, nullptr, nullptr, H, 1.0f);

  // Split/cast q,k and transpose v.
  qkv_split_kernel<<<(S * H) / 256, blk, 0, stream>>>(qkvf, qb, kb, vT, S, H, HD);

  // Attention per head: scores -> causal softmax -> ctx.
  // (16 MB scores + 8 MB probs stay resident in the 192 MB L2.)
  for (int h = 0; h < NH; ++h) {
    const bf16* qh = qb + (size_t)h * S * HD;
    const bf16* kh = kb + (size_t)h * S * HD;
    const bf16* vh = vT + (size_t)h * HD * S;
    gemm_bf16_wmma_kernel<0><<<dim3(S / BN, S / BM), blk, 0, stream>>>(
        qh, HD, kh, HD, scoresb, nullptr, S, nullptr, nullptr, HD, SCALE);
    softmax_causal_kernel<<<S, blk, 0, stream>>>(scoresb, probsb, S);
    gemm_bf16_wmma_kernel<1><<<dim3(HD / BN, S / BM), blk, 0, stream>>>(
        probsb, S, vh, S, nullptr, ctx_bf + (size_t)h * HD, H, nullptr, nullptr, S, 1.0f);
  }

  // x1 = x + ctx @ w_attn_out^T   (residual fused in epilogue)
  gemm_bf16_wmma_kernel<2><<<dim3(H / BN, S / BM), blk, 0, stream>>>(
      ctx_bf, H, wattn_bf, H, x1, nullptr, H, hidden, nullptr, H, 1.0f);

  // ln2 + gated MLP.
  rmsnorm_bf16_kernel<<<S, blk, 0, stream>>>(x1, w_norm2, ln_bf, H);
  gemm_bf16_wmma_kernel<0><<<dim3(FF2 / BN, S / BM), blk, 0, stream>>>(
      ln_bf, H, w_h4h_bf, H, inter, nullptr, FF2, nullptr, nullptr, H, 1.0f);
  swiglu_kernel<<<8192, blk, 0, stream>>>(inter, act_bf, S, FF);

  // out = x1 + bias + act @ w_deq^T
  gemm_bf16_wmma_kernel<3><<<dim3(H / BN, S / BM), blk, 0, stream>>>(
      act_bf, FF, wdeq_bf, FF, out, nullptr, H, x1, bias_4h, FF, 1.0f);
}
